// BaselineGIN_42502996361221
// MI455X (gfx1250) — compile-verified, matrix operand order checked by reading, not run on hardware
//
#include <hip/hip_runtime.h>

// ---------------------------------------------------------------------------
// GIN forward for MI455X (gfx1250, wave32).
//  - GEMMs: v_wmma_f32_16x16x32_f16, f16 inputs / f32 accumulate
//  - BN affine fused into all activation consumers (no standalone BN pass)
//  - BN batch statistics fused into GEMM epilogues (no colstats pass)
//  - weights pre-converted to f16 and pre-transposed (K-major) once
//  - weight slab staging uses GLOBAL_LOAD_ASYNC_TO_LDS_B128 when available
// ---------------------------------------------------------------------------

typedef __attribute__((ext_vector_type(16))) _Float16 v16h;
typedef __attribute__((ext_vector_type(8)))  float    v8f;
typedef __attribute__((ext_vector_type(4)))  int      v4i;
typedef __attribute__((address_space(1)))    v4i      gv4i;  // global int4
typedef __attribute__((address_space(3)))    v4i      lv4i;  // LDS int4

#define DF 128      // feature dim
#define C1 256      // hidden dim (2*DF)
#define NG 512      // num graphs (reference constant)

#if defined(__has_builtin)
#if __has_builtin(__builtin_amdgcn_global_load_async_to_lds_b128) && \
    __has_builtin(__builtin_amdgcn_s_wait_asynccnt)
#define USE_ASYNC_LDS 1
#endif
#endif
#ifndef USE_ASYNC_LDS
#define USE_ASYNC_LDS 0
#endif

// Build a WMMA 16x32 f16 fragment from LDS stored [outer][k] row-major
// (A: outer=M rows; B: outer=N cols of K-major W^T). ISA layout:
// lane%16 -> outer; lane/16 selects K-halves; pairs contiguous => 2x b128.
__device__ __forceinline__ v16h frag_ld(const _Float16* lds, int stride,
                                        int outer0, int k0) {
  const int lane = threadIdx.x & 31;
  const int o  = outer0 + (lane & 15);
  const int hf = (lane >> 4) << 3;
  const _Float16* p = lds + o * stride + k0;
  union { v16h v; unsigned u[8]; } f;
#pragma unroll
  for (int r = 0; r < 8; ++r) {
    const int kk = (r < 4) ? (r * 2 + hf) : (16 + (r - 4) * 2 + hf);
    f.u[r] = *reinterpret_cast<const unsigned*>(p + kk);
  }
  return f.v;
}

__device__ __forceinline__ v8f wmma_f16(v16h a, v16h b, v8f c) {
  return __builtin_amdgcn_wmma_f32_16x16x32_f16(false, a, false, b,
                                                (short)0, c, false, false);
}

// Copy an (ncols x 32) f16 slab of K-major W^T into LDS (dest stride 40).
// 16-byte chunks; async-to-LDS path when the toolchain exposes it.
__device__ __forceinline__ void stage_wslab(const _Float16* __restrict__ Wt,
                                            int Krow, _Float16* sW, int ncols,
                                            int k0, int tid) {
  const int chunks = ncols * 4;
  for (int idx = tid; idx < chunks; idx += 256) {
    int nn = idx >> 2, c = idx & 3;
    const _Float16* g = Wt + (long)nn * Krow + k0 + c * 8;
    _Float16* l = sW + nn * 40 + c * 8;
#if USE_ASYNC_LDS
    __builtin_amdgcn_global_load_async_to_lds_b128((gv4i*)g, (lv4i*)l, 0, 0);
#else
    *reinterpret_cast<uint4*>(l) = *reinterpret_cast<const uint4*>(g);
#endif
  }
#if USE_ASYNC_LDS
  __builtin_amdgcn_s_wait_asynccnt(0);
#endif
}

// --------------------------- small helpers ---------------------------------

__global__ void k_zero(float* __restrict__ p, long n) {
  long i = (long)blockIdx.x * blockDim.x + threadIdx.x;
  if (i < n) p[i] = 0.f;
}

// f16 + transposed (K-major) copies of all layer weights
__global__ void k_prepw(const float* __restrict__ W1, const float* __restrict__ W2,
                        _Float16* __restrict__ Wt1, _Float16* __restrict__ Wt2,
                        int L) {
  long i = (long)blockIdx.x * blockDim.x + threadIdx.x;
  const long per = (long)DF * C1;
  const long tot = (long)L * per;
  if (i < tot) {                       // Wt1[l][n<256][k<128] = W1[l][k][n]
    int l = (int)(i / per), e = (int)(i % per);
    int n = e / DF, k = e % DF;
    Wt1[i] = (_Float16)W1[(long)l * per + (long)k * C1 + n];
  } else if (i < 2 * tot) {            // Wt2[l][n<128][k<256] = W2[l][k][n]
    long j = i - tot;
    int l = (int)(j / per), e = (int)(j % per);
    int n = e / C1, k = e % C1;
    Wt2[j] = (_Float16)W2[(long)l * per + (long)k * DF + n];
  }
}

// hagg = (1+eps_l) * t(x);  t(v) = apply ? relu(v*sc[c]+sh[c]) : v
__global__ void k_init(const float* __restrict__ x, float* __restrict__ hagg,
                       const float* __restrict__ epsl,
                       const float* __restrict__ sc, const float* __restrict__ sh,
                       int apply, long n) {
  long i = (long)blockIdx.x * blockDim.x + threadIdx.x;
  if (i >= n) return;
  float v = x[i];
  if (apply) {
    int c = (int)(i % DF);
    v = fmaxf(fmaf(v, sc[c], sh[c]), 0.f);
  }
  hagg[i] = (1.0f + epsl[0]) * v;
}

// scatter-add t(x[src]) into hagg[dst]; one wave32 per edge
__global__ __launch_bounds__(256) void k_edges(const int* __restrict__ src,
                                               const int* __restrict__ dst,
                                               const float* __restrict__ x,
                                               const float* __restrict__ sc,
                                               const float* __restrict__ sh,
                                               int apply,
                                               float* __restrict__ hagg, int E) {
  int e = (int)(((long)blockIdx.x * blockDim.x + threadIdx.x) >> 5);
  int lane = threadIdx.x & 31;
  if (e >= E) return;
  int s = src[e], d = dst[e];
  float4 v = *reinterpret_cast<const float4*>(x + (long)s * DF + lane * 4);
  if (apply) {
    int c = lane * 4;
    v.x = fmaxf(fmaf(v.x, sc[c + 0], sh[c + 0]), 0.f);
    v.y = fmaxf(fmaf(v.y, sc[c + 1], sh[c + 1]), 0.f);
    v.z = fmaxf(fmaf(v.z, sc[c + 2], sh[c + 2]), 0.f);
    v.w = fmaxf(fmaf(v.w, sc[c + 3], sh[c + 3]), 0.f);
  }
  float* o = hagg + (long)d * DF + lane * 4;
  atomicAdd(o + 0, v.x); atomicAdd(o + 1, v.y);
  atomicAdd(o + 2, v.z); atomicAdd(o + 3, v.w);
}

// scale = g*rsqrt(var+eps); shift = beta - mean*scale
__global__ void k_finalize(const float* __restrict__ stats,
                           const float* __restrict__ g,
                           const float* __restrict__ be,
                           float* __restrict__ scale, float* __restrict__ shift,
                           int C, float inv_rows) {
  int c = threadIdx.x;
  if (c >= C) return;
  float mean = stats[c] * inv_rows;
  float var  = stats[C + c] * inv_rows - mean * mean;
  float rs   = rsqrtf(var + 1e-5f);
  float scv  = g[c] * rs;
  scale[c] = scv;
  shift[c] = be[c] - mean * scv;
}

// column stats over rows (classifier BN only)
__global__ void k_colstats(const float* __restrict__ P, float* __restrict__ stats,
                           int C, int rows) {
  int c = threadIdx.x;                 // blockDim.x == C, single block
  float s = 0.f, q = 0.f;
  for (int r = 0; r < rows; ++r) {
    float v = P[(long)r * C + c];
    s += v; q += v * v;
  }
  atomicAdd(&stats[c], s);
  atomicAdd(&stats[C + c], q);
}

__global__ void k_bnrelu(const float* __restrict__ in, float* __restrict__ out,
                         const float* __restrict__ scale,
                         const float* __restrict__ shift, int C, long n) {
  long i = (long)blockIdx.x * blockDim.x + threadIdx.x;
  if (i >= n) return;
  int c = (int)(i % C);
  out[i] = fmaxf(fmaf(in[i], scale[c], shift[c]), 0.f);
}

// ------------------------------- GEMM 1 ------------------------------------
// Y[N,256](f16) = f16(H[N,128]) @ Wt1^T + b ; fused column sum/sumsq.
__global__ __launch_bounds__(256) void k_gemm1(const float* __restrict__ H,
                                               const _Float16* __restrict__ Wt,
                                               const float* __restrict__ bias,
                                               _Float16* __restrict__ Y,
                                               float* __restrict__ stats, int n) {
  __shared__ _Float16 sA[64 * 136];   // 64 rows x 128 K (pad 8)
  __shared__ _Float16 sW[C1 * 40];    // 256 cols x 32-K slab, K-major (pad 8)
  const int tid = threadIdx.x;
  const int r0  = blockIdx.x * 64;

  for (int idx = tid; idx < 2048; idx += 256) {       // stage A (f32 -> f16)
    int e = idx << 2, rr = e >> 7, kk = e & 127;
    int row = r0 + rr;
    float4 a = (row < n) ? *reinterpret_cast<const float4*>(H + (long)row * DF + kk)
                         : make_float4(0.f, 0.f, 0.f, 0.f);
    _Float16* d = sA + rr * 136 + kk;
    d[0] = (_Float16)a.x; d[1] = (_Float16)a.y;
    d[2] = (_Float16)a.z; d[3] = (_Float16)a.w;
  }

  const int wave = tid >> 5, wr = wave & 1, wc = wave >> 1;
  const v8f vz = {0.f, 0.f, 0.f, 0.f, 0.f, 0.f, 0.f, 0.f};
  v8f acc[2][4];
#pragma unroll
  for (int i = 0; i < 2; ++i)
#pragma unroll
    for (int j = 0; j < 4; ++j) acc[i][j] = vz;

  for (int k0 = 0; k0 < DF; k0 += 32) {
    if (k0 + 32 < DF) __builtin_prefetch(Wt + (k0 + 32), 0, 0);
    __syncthreads();
    stage_wslab(Wt, DF, sW, C1, k0, tid);
    __syncthreads();
    v16h a0 = frag_ld(sA, 136, wr * 32 + 0,  k0);
    v16h a1 = frag_ld(sA, 136, wr * 32 + 16, k0);
#pragma unroll
    for (int tn = 0; tn < 4; ++tn) {
      v16h bf = frag_ld(sW, 40, wc * 64 + tn * 16, 0);
      acc[0][tn] = wmma_f16(a0, bf, acc[0][tn]);
      acc[1][tn] = wmma_f16(a1, bf, acc[1][tn]);
    }
  }

  const int lane = tid & 31, nl = lane & 15, mo = (lane >> 4) * 8;
#pragma unroll
  for (int tn = 0; tn < 4; ++tn) {
    int col = wc * 64 + tn * 16 + nl;
    float bv = bias[col];
    float s = 0.f, q = 0.f;
#pragma unroll
    for (int tm = 0; tm < 2; ++tm) {
#pragma unroll
      for (int r = 0; r < 8; ++r) {
        int row = r0 + wr * 32 + tm * 16 + mo + r;
        if (row < n) {
          float v = acc[tm][tn][r] + bv;
          Y[(long)row * C1 + col] = (_Float16)v;
          s += v; q += v * v;
        }
      }
    }
    atomicAdd(&stats[col], s);
    atomicAdd(&stats[C1 + col], q);
  }
}

// ------------------------------- GEMM 2 ------------------------------------
// H2[N,128](f32) = relu(Y*scale+shift)(f16) @ Wt2^T + b ; fused stats.
__global__ __launch_bounds__(256) void k_gemm2(const _Float16* __restrict__ Y,
                                               const float* __restrict__ scale,
                                               const float* __restrict__ shift,
                                               const _Float16* __restrict__ Wt,
                                               const float* __restrict__ bias,
                                               float* __restrict__ H2,
                                               float* __restrict__ stats, int n) {
  __shared__ _Float16 sA[64 * 264];   // 64 rows x 256 K (pad 8)
  __shared__ _Float16 sW[DF * 40];    // 128 cols x 32-K slab, K-major
  const int tid = threadIdx.x;
  const int r0  = blockIdx.x * 64;

  for (int idx = tid; idx < 4096; idx += 256) {       // stage A: BN+ReLU fold
    int e = idx << 2, rr = e >> 8, kk = e & 255;
    int row = r0 + rr;
    _Float16* d = sA + rr * 264 + kk;
    if (row < n) {
      const _Float16* yp = Y + (long)row * C1 + kk;
#pragma unroll
      for (int j = 0; j < 4; ++j) {
        float v = fmaf((float)yp[j], scale[kk + j], shift[kk + j]);
        d[j] = (_Float16)fmaxf(v, 0.f);
      }
    } else {
      d[0] = d[1] = d[2] = d[3] = (_Float16)0.f;
    }
  }

  const int wave = tid >> 5, wr = wave & 3, wc = wave >> 2;
  const v8f vz = {0.f, 0.f, 0.f, 0.f, 0.f, 0.f, 0.f, 0.f};
  v8f acc[4] = {vz, vz, vz, vz};

  for (int k0 = 0; k0 < C1; k0 += 32) {
    if (k0 + 32 < C1) __builtin_prefetch(Wt + (k0 + 32), 0, 0);
    __syncthreads();
    stage_wslab(Wt, C1, sW, DF, k0, tid);
    __syncthreads();
    v16h a = frag_ld(sA, 264, wr * 16, k0);
#pragma unroll
    for (int tn = 0; tn < 4; ++tn) {
      v16h bf = frag_ld(sW, 40, wc * 64 + tn * 16, 0);
      acc[tn] = wmma_f16(a, bf, acc[tn]);
    }
  }

  const int lane = tid & 31, nl = lane & 15, mo = (lane >> 4) * 8;
#pragma unroll
  for (int tn = 0; tn < 4; ++tn) {
    int col = wc * 64 + tn * 16 + nl;
    float bv = bias[col];
    float s = 0.f, q = 0.f;
#pragma unroll
    for (int r = 0; r < 8; ++r) {
      int row = r0 + wr * 16 + mo + r;
      if (row < n) {
        float v = acc[tn][r] + bv;
        H2[(long)row * DF + col] = v;
        s += v; q += v * v;
      }
    }
    atomicAdd(&stats[col], s);
    atomicAdd(&stats[DF + col], q);
  }
}

// ------------------------------ readout ------------------------------------

// one wave32 per node; pools t(x) (>=0 after relu => int-bits atomicMax ok)
__global__ __launch_bounds__(256) void k_pool(const float* __restrict__ x,
                                              const int* __restrict__ batch,
                                              const float* __restrict__ sc,
                                              const float* __restrict__ sh,
                                              float* __restrict__ psum,
                                              int* __restrict__ pmax,
                                              float* __restrict__ pcnt, int n) {
  int node = (int)(((long)blockIdx.x * blockDim.x + threadIdx.x) >> 5);
  int lane = threadIdx.x & 31;
  if (node >= n) return;
  int g = batch[node];
  float4 v = *reinterpret_cast<const float4*>(x + (long)node * DF + lane * 4);
  int c = lane * 4;
  v.x = fmaxf(fmaf(v.x, sc[c + 0], sh[c + 0]), 0.f);
  v.y = fmaxf(fmaf(v.y, sc[c + 1], sh[c + 1]), 0.f);
  v.z = fmaxf(fmaf(v.z, sc[c + 2], sh[c + 2]), 0.f);
  v.w = fmaxf(fmaf(v.w, sc[c + 3], sh[c + 3]), 0.f);
  float* s = psum + (long)g * DF + c;
  atomicAdd(s + 0, v.x); atomicAdd(s + 1, v.y);
  atomicAdd(s + 2, v.z); atomicAdd(s + 3, v.w);
  int* m = pmax + (long)g * DF + c;
  atomicMax(m + 0, __float_as_int(v.x)); atomicMax(m + 1, __float_as_int(v.y));
  atomicMax(m + 2, __float_as_int(v.z)); atomicMax(m + 3, __float_as_int(v.w));
  if (lane == 0) atomicAdd(&pcnt[g], 1.0f);
}

__global__ void k_buildz(const float* __restrict__ psum,
                         const int* __restrict__ pmax,
                         const float* __restrict__ pcnt,
                         float* __restrict__ z, int G) {
  int idx = blockIdx.x * blockDim.x + threadIdx.x;
  if (idx >= G * DF) return;
  int g = idx / DF, c = idx % DF;
  float s = psum[idx];
  float cnt = fmaxf(pcnt[g], 1.0f);
  z[(long)g * (3 * DF) + c]          = s;
  z[(long)g * (3 * DF) + DF + c]     = s / cnt;
  z[(long)g * (3 * DF) + 2 * DF + c] = __int_as_float(pmax[idx]);
}

__global__ void k_dense(const float* __restrict__ A, const float* __restrict__ W,
                        const float* __restrict__ b, float* __restrict__ O,
                        int rows, int K, int C, int relu) {
  int idx = blockIdx.x * blockDim.x + threadIdx.x;
  if (idx >= rows * C) return;
  int r = idx / C, c = idx % C;
  float acc = b[c];
  for (int k = 0; k < K; ++k) acc = fmaf(A[(long)r * K + k], W[(long)k * C + c], acc);
  O[idx] = relu ? fmaxf(acc, 0.f) : acc;
}

// ---------------------------------------------------------------------------

extern "C" void kernel_launch(void* const* d_in, const int* in_sizes, int n_in,
                              void* d_out, int out_size, void* d_ws, size_t ws_size,
                              hipStream_t stream) {
  const int N = in_sizes[0] / DF;
  const int E = in_sizes[1] / 2;
  const int L = in_sizes[12];

  const float* x_in  = (const float*)d_in[0];
  const int*   ei    = (const int*)d_in[1];
  const int*   batch = (const int*)d_in[2];
  const float* W1  = (const float*)d_in[4];
  const float* b1  = (const float*)d_in[5];
  const float* g1  = (const float*)d_in[6];
  const float* be1 = (const float*)d_in[7];
  const float* W2  = (const float*)d_in[8];
  const float* b2  = (const float*)d_in[9];
  const float* gbn = (const float*)d_in[10];
  const float* bbn = (const float*)d_in[11];
  const float* eps = (const float*)d_in[12];
  const float* cW1 = (const float*)d_in[13];
  const float* cb1 = (const float*)d_in[14];
  const float* cg  = (const float*)d_in[15];
  const float* cbt = (const float*)d_in[16];
  const float* cW2 = (const float*)d_in[17];
  const float* cb2 = (const float*)d_in[18];
  const float* cW3 = (const float*)d_in[19];
  const float* cb3 = (const float*)d_in[20];
  float* out = (float*)d_out;

  const long per = (long)DF * C1;      // 32768 weights per layer per matrix

  // workspace carve-out
  char* w = (char*)d_ws;
  float*    A     = (float*)w;     w += (size_t)N * DF * 4;   // xsrc / H2 ping
  float*    hagg  = (float*)w;     w += (size_t)N * DF * 4;
  _Float16* yf16  = (_Float16*)w;  w += (size_t)N * C1 * 2;
  _Float16* Wt1   = (_Float16*)w;  w += (size_t)L * per * 2;
  _Float16* Wt2   = (_Float16*)w;  w += (size_t)L * per * 2;
  float*    stats = (float*)w;     w += 768 * 4;
  float*    scA   = (float*)w;     w += 256 * 4;
  float*    shA   = (float*)w;     w += 256 * 4;
  float*    scB   = (float*)w;     w += 128 * 4;
  float*    shB   = (float*)w;     w += 128 * 4;
  float*    psum  = (float*)w;     w += (size_t)NG * DF * 4;
  int*      pmax  = (int*)w;       w += (size_t)NG * DF * 4;
  float*    pcnt  = (float*)w;     w += (size_t)NG * 4;
  float*    z     = (float*)w;     w += (size_t)NG * 3 * DF * 4;
  float*    z1    = (float*)w;     w += (size_t)NG * DF * 4;
  float*    z2    = (float*)w;     w += (size_t)NG * 64 * 4;

  const long totD  = (long)N * DF;
  const int  ebD   = (int)((totD + 255) / 256);
  const int  nbG   = (N + 63) / 64;
  const int  ebE   = (E + 7) / 8;
  const float invN = 1.0f / (float)N;

  const int* src = ei;
  const int* dst = ei + E;

  k_prepw<<<(int)((2 * L * per + 255) / 256), 256, 0, stream>>>(W1, W2, Wt1, Wt2, L);

  const float* xsrc = x_in;
  for (int l = 0; l < L; ++l) {
    int apply = (l > 0);
    k_init<<<ebD, 256, 0, stream>>>(xsrc, hagg, eps + l, scB, shB, apply, totD);
    k_edges<<<ebE, 256, 0, stream>>>(src, dst, xsrc, scB, shB, apply, hagg, E);
    k_zero<<<3, 256, 0, stream>>>(stats, 768);
    k_gemm1<<<nbG, 256, 0, stream>>>(hagg, Wt1 + (size_t)l * per,
                                     b1 + (size_t)l * C1, yf16, stats, N);
    k_finalize<<<1, C1, 0, stream>>>(stats, g1 + (size_t)l * C1,
                                     be1 + (size_t)l * C1, scA, shA, C1, invN);
    k_zero<<<3, 256, 0, stream>>>(stats, 768);
    k_gemm2<<<nbG, 256, 0, stream>>>(yf16, scA, shA, Wt2 + (size_t)l * per,
                                     b2 + (size_t)l * DF, A, stats, N);
    k_finalize<<<1, DF, 0, stream>>>(stats, gbn + (size_t)l * DF,
                                     bbn + (size_t)l * DF, scB, shB, DF, invN);
    xsrc = A;
  }

  // --- multi-pool readout + classifier ---
  k_zero<<<(NG * (2 * DF + 1) + 255) / 256, 256, 0, stream>>>(psum,
                                                              (long)NG * (2 * DF + 1));
  k_pool<<<(N + 7) / 8, 256, 0, stream>>>(xsrc, batch, scB, shB, psum, pmax, pcnt, N);
  k_buildz<<<(NG * DF + 255) / 256, 256, 0, stream>>>(psum, pmax, pcnt, z, NG);

  k_dense<<<(NG * DF + 255) / 256, 256, 0, stream>>>(z, cW1, cb1, z1, NG, 3 * DF, DF, 0);
  k_zero<<<3, 256, 0, stream>>>(stats, 768);
  k_colstats<<<1, DF, 0, stream>>>(z1, stats, DF, NG);
  k_finalize<<<1, DF, 0, stream>>>(stats, cg, cbt, scA, shA, DF, 1.0f / (float)NG);
  k_bnrelu<<<(NG * DF + 255) / 256, 256, 0, stream>>>(z1, z1, scA, shA, DF,
                                                      (long)NG * DF);
  k_dense<<<(NG * 64 + 255) / 256, 256, 0, stream>>>(z1, cW2, cb2, z2, NG, DF, 64, 1);
  k_dense<<<(NG * 2 + 255) / 256, 256, 0, stream>>>(z2, cW3, cb3, out, NG, 64, 2, 0);

  (void)n_in; (void)out_size; (void)ws_size;
}
